// RNN_torch_53738630807663
// MI455X (gfx1250) — matrix operand order repaired and checked by reading, not compiled, not run on hardware
//
#include <hip/hip_runtime.h>
#include <hip/hip_bf16.h>
#include <math.h>

#define T_STEPS 1200
#define BATCH   128
#define NN      512
#define INSZ    6
#define KP      544          // 512 (rec) + 6 (inp) padded to 17 K-tiles of 32
#define APITCH  544
#define ALPHA_F 0.1f

typedef __bf16 bf16;
typedef __attribute__((ext_vector_type(16))) __bf16 bf16x16;
typedef __attribute__((ext_vector_type(8)))  __bf16 bf16x8;
typedef __attribute__((ext_vector_type(8)))  float  f32x8;

union BV16 { bf16x16 v; bf16x8 h[2]; };

__device__ __forceinline__ bf16 to_bf16(float f) {
    unsigned x = __builtin_bit_cast(unsigned, f);
    unsigned r = (x + 0x7FFFu + ((x >> 16) & 1u)) >> 16;   // round-to-nearest-even
    unsigned short s = (unsigned short)r;
    return __builtin_bit_cast(bf16, s);
}

// ---------------- kernel 1: pack [W_rec | W_inp | 0] -> bf16 Wc[NN][KP] ----------------
__global__ void pack_weights(const float* __restrict__ W_rec,
                             const float* __restrict__ W_inp,
                             bf16* __restrict__ Wc) {
    int idx = blockIdx.x * blockDim.x + threadIdx.x;
    if (idx >= NN * KP) return;
    int n = idx / KP, k = idx - n * KP;
    float v = 0.0f;
    if (k < NN)             v = W_rec[n * NN + k];
    else if (k < NN + INSZ) v = W_inp[n * INSZ + (k - NN)];
    Wc[idx] = to_bf16(v);
}

// ---------------- kernel 2: persistent recurrence, one WG per 16 batch rows ------------
__global__ __launch_bounds__(512)
void rnn_persistent(const bf16* __restrict__ Wc,
                    const float* __restrict__ u,
                    const float* __restrict__ y_init,
                    const float* __restrict__ rec_noise,
                    const float* __restrict__ inp_noise,
                    float* __restrict__ states) {
    __shared__ __align__(16) bf16  Abuf[16 * APITCH];  // A = [s_bf16 | u+noise | pad]
    __shared__ __align__(16) float preLds[16 * NN];    // pre-activation scratch

    const int tid  = threadIdx.x;           // 512 threads = 16 waves
    const int b0   = blockIdx.x * 16;       // batch slice
    const int lane = tid & 31;
    const int wv   = tid >> 5;              // wave id 0..15 -> owns N-tiles 2wv, 2wv+1
    const int m    = lane & 15;
    const int h    = lane >> 4;

    // ---- prologue: s0 = y_init broadcast; write states[:,0,:]; zero A pad cols ----
    float sreg[16];                         // f32 state: this thread owns column n=tid
    {
        const int n = tid;
        const float v0 = y_init[n];
        #pragma unroll
        for (int b = 0; b < 16; ++b) {
            sreg[b] = v0;
            Abuf[b * APITCH + n] = to_bf16(v0);
        }
        float4 v4 = make_float4(v0, v0, v0, v0);
        float* dst = states + (size_t)n * T_STEPS * BATCH + b0;
        *(float4*)(dst + 0)  = v4;  *(float4*)(dst + 4)  = v4;
        *(float4*)(dst + 8)  = v4;  *(float4*)(dst + 12) = v4;
    }
    if (tid < 16 * 32) {                    // columns 512..543 (inputs rewritten per step)
        int b = tid >> 5, c = NN + (tid & 31);
        Abuf[b * APITCH + c] = to_bf16(0.0f);
    }
    __syncthreads();

    // fragment base pointers
    const bf16* aBase  = &Abuf[m * APITCH + 16 * h];
    const int   nn     = 2 * wv * 16 + m;                       // acc0 column
    const bf16* bBase0 = Wc + (size_t)nn * KP + 16 * h;         // B frag: contiguous 32B
    const bf16* bBase1 = bBase0 + (size_t)16 * KP;

    for (int t = 0; t < T_STEPS - 1; ++t) {
        // ---- fill input columns of A for this step: A[b][512+i] = u_t + inp_noise_t ----
        if (tid < 96) {
            int i = tid >> 4, bb = tid & 15;
            size_t off = ((size_t)i * T_STEPS + t) * BATCH + b0 + bb;
            Abuf[bb * APITCH + NN + i] = to_bf16(u[off] + inp_noise[off]);
        }
        __syncthreads();

        // Defeat LICM on the weight loads without losing the global address space:
        // an opaque, per-iteration zero offset keeps the 34 B fragments streaming
        // from the L2-resident Wc as global_load_b128 (SGPR base + VGPR offset)
        // instead of being hoisted (272 VGPRs/wave) and spilled to scratch, and
        // keeps them off the flat path (no LDS-aperture probe, no DScnt coupling).
        int zoff = 0;
        asm volatile("" : "+v"(zoff));
        const bf16* b0p = bBase0 + zoff;
        const bf16* b1p = bBase1 + zoff;

        // ---- GEMM: pre = A(16xKP) * Wc^T, two 16x16 N-tiles per wave ----
        f32x8 acc0 = {};
        f32x8 acc1 = {};
        #pragma unroll
        for (int kt = 0; kt < KP / 32; ++kt) {
            BV16 a, bA, bB;
            a.h[0] = *(const bf16x8*)(aBase + kt * 32);         // K = k0+16h .. +7
            a.h[1] = *(const bf16x8*)(aBase + kt * 32 + 8);     // K = k0+16h+8 .. +15
            bA.v   = *(const bf16x16*)(b0p + kt * 32);          // Wc[n][k0+16h .. +15]
            bB.v   = *(const bf16x16*)(b1p + kt * 32);
            acc0 = __builtin_amdgcn_wmma_f32_16x16x32_bf16(false, a.v, false, bA.v,
                                                           (short)0, acc0, false, false);
            acc1 = __builtin_amdgcn_wmma_f32_16x16x32_bf16(false, a.v, false, bB.v,
                                                           (short)0, acc1, false, false);
        }
        // D layout: VGPR v -> row (v + 8h), col = lane&15 (+ tile base)
        #pragma unroll
        for (int v = 0; v < 8; ++v) {
            preLds[(v + 8 * h) * NN + nn]      = acc0[v];
            preLds[(v + 8 * h) * NN + nn + 16] = acc1[v];
        }
        __syncthreads();

        // ---- update: thread owns column n=tid, all 16 batch rows ----
        {
            const int n = tid;
            const float* rn = rec_noise + ((size_t)n * T_STEPS + t) * BATCH + b0;
            float4 r[4];
            r[0] = *(const float4*)(rn + 0);
            r[1] = *(const float4*)(rn + 4);
            r[2] = *(const float4*)(rn + 8);
            r[3] = *(const float4*)(rn + 12);
            const float* rf = (const float*)r;
            float outv[16];
            #pragma unroll
            for (int b = 0; b < 16; ++b) {
                float pr = preLds[b * NN + n];
                float sn = (1.0f - ALPHA_F) * sreg[b] + ALPHA_F * (tanhf(pr) + rf[b]);
                sreg[b] = sn;
                Abuf[b * APITCH + n] = to_bf16(sn);
                outv[b] = sn;
            }
            float* dst = states + ((size_t)n * T_STEPS + (t + 1)) * BATCH + b0;
            *(float4*)(dst + 0)  = make_float4(outv[0],  outv[1],  outv[2],  outv[3]);
            *(float4*)(dst + 4)  = make_float4(outv[4],  outv[5],  outv[6],  outv[7]);
            *(float4*)(dst + 8)  = make_float4(outv[8],  outv[9],  outv[10], outv[11]);
            *(float4*)(dst + 12) = make_float4(outv[12], outv[13], outv[14], outv[15]);
            if (t + 2 < T_STEPS)   // pull next step's noise line into cache
                __builtin_prefetch(rec_noise + ((size_t)n * T_STEPS + t + 1) * BATCH + b0, 0, 0);
        }
        // no barrier needed here: next iteration's post-fill barrier covers A reuse
    }
}

// ---------------- kernel 3: outputs = states @ W_out^T, coalesced over batch ----------
__global__ void out_proj(const float* __restrict__ states,
                         const float* __restrict__ W_out,
                         float* __restrict__ outs) {
    const int t = blockIdx.x;       // 0..1199
    const int b = threadIdx.x;      // 0..127
    const float* sp = states + (size_t)t * BATCH + b;
    float a0 = 0.f, a1 = 0.f;
    #pragma unroll 8
    for (int n = 0; n < NN; ++n) {
        float sv = sp[(size_t)n * T_STEPS * BATCH];
        a0 = fmaf(W_out[n],      sv, a0);
        a1 = fmaf(W_out[NN + n], sv, a1);
    }
    outs[(size_t)t * BATCH + b]                           = a0;
    outs[(size_t)T_STEPS * BATCH + (size_t)t * BATCH + b] = a1;
}

extern "C" void kernel_launch(void* const* d_in, const int* in_sizes, int n_in,
                              void* d_out, int out_size, void* d_ws, size_t ws_size,
                              hipStream_t stream) {
    const float* u         = (const float*)d_in[0];
    const float* W_rec     = (const float*)d_in[1];
    const float* W_inp     = (const float*)d_in[2];
    const float* W_out     = (const float*)d_in[3];
    const float* y_init    = (const float*)d_in[4];
    const float* rec_noise = (const float*)d_in[5];
    const float* inp_noise = (const float*)d_in[6];

    float* states = (float*)d_out;                                   // (N,T,B)
    float* outs   = (float*)d_out + (size_t)NN * T_STEPS * BATCH;    // (2,T,B)
    bf16*  Wc     = (bf16*)d_ws;                                     // 512x544 bf16

    pack_weights<<<(NN * KP + 255) / 256, 256, 0, stream>>>(W_rec, W_inp, Wc);
    rnn_persistent<<<BATCH / 16, 512, 0, stream>>>(Wc, u, y_init, rec_noise,
                                                   inp_noise, states);
    out_proj<<<T_STEPS, BATCH, 0, stream>>>(states, W_out, outs);
}
